// AEDGAT_layer_24489903522520
// MI455X (gfx1250) — matrix operand order repaired and checked by reading, not compiled, not run on hardware
//
#include <hip/hip_runtime.h>
#include <hip/hip_bf16.h>

// ---------------- problem constants (match reference) ----------------
#define H_  8
#define HD_ 128
#define NT_ 16000
#define NQ_ 4000
#define B_  64
#define ET_ 256000
#define EQ_ 64000

typedef __attribute__((ext_vector_type(16))) __bf16 v16bf;
typedef __attribute__((ext_vector_type(8)))  float  v8f;

// ---------------- helpers ----------------
__device__ __forceinline__ unsigned bfbits(float f) {
    union { float f; unsigned u; } x; x.f = f;
    unsigned r = x.u + 0x7FFFu + ((x.u >> 16) & 1u);   // round-to-nearest-even
    return r >> 16;
}
__device__ __forceinline__ __bf16 bfval(float f) {
    union { unsigned short s; __bf16 b; } y;
    y.s = (unsigned short)bfbits(f);
    return y.b;
}

__device__ __forceinline__ float atomicMaxFloat(float* addr, float val) {
    int* ai = (int*)addr;
    int old = __float_as_int(*addr);
    while (__int_as_float(old) < val) {
        int assumed = old;
        old = atomicCAS(ai, assumed, __float_as_int(val));
        if (old == assumed) break;
    }
    return __int_as_float(old);
}

// ---------------- generic bf16 WMMA GEMM: C = scale*(A@B) (+bias) ----------------
// A: [M,K] row-major (lda). B: [K,N] row-major (ldb), or transB: B is [N,K] row-major.
// 64x128 block tile, BK=32, 256 threads = 8 waves.
// Wave w: A row-tile (w&3), B col-tiles 4*(w>>2) .. +3  -> 4 WMMAs / wave / K-step.
// LDS tiles are in WMMA *fragment layout* -> fragment loads are ds_load_b128 pairs.
//   A frag:              element (r, cc): lane=(cc>=16)*16+r,  e=cc&15
//   B frag:              element (kk, c): lane=(kk>=16)*16+c,   e=kk&15
#define BM 64
#define BN 128
#define BK 32

__global__ __launch_bounds__(256)
void gemm_bf16_wmma(const float* __restrict__ A, const float* __restrict__ B,
                    const float* __restrict__ bias, float* __restrict__ C,
                    int M, int N, int K, int lda, int ldb, int ldc,
                    int transB, float scale)
{
    __shared__ __bf16 Asf[4][32][16];   // [row-tile][lane][e]  4KB
    __shared__ __bf16 Bsf[8][32][16];   // [col-tile][lane][e]  8KB

    const int tid  = threadIdx.x;
    const int wave = tid >> 5;
    const int lane = tid & 31;
    const int bm = blockIdx.y * BM;
    const int bn = blockIdx.x * BN;

    const int art  = wave & 3;          // A row tile (16 rows each)
    const int bct0 = (wave >> 2) * 4;   // B col tiles bct0 .. bct0+3

    const bool mfull = (bm + BM <= M);
    const bool nfull = (bn + BN <= N);

    v8f acc[4] = {{}, {}, {}, {}};

    for (int k0 = 0; k0 < K; k0 += BK) {
        const bool fast = mfull && nfull && (k0 + BK <= K);   // uniform

        if (fast) {
            // ---- A: 64x32, 1024 pairs along K: contiguous float2, one ds_store_b32
#pragma unroll
            for (int it = 0; it < 4; ++it) {
                int t  = tid + it * 256;
                int rr = t >> 4;
                int cc = (t & 15) << 1;
                const float* p = A + (long)(bm + rr) * lda + (k0 + cc);
                float v0 = p[0], v1 = p[1];
                int tile = rr >> 4, r = rr & 15;
                int flane = ((cc >> 4) & 1) * 16 + r;
                int e     = cc & 15;
                *(unsigned*)&Asf[tile][flane][e] = bfbits(v0) | (bfbits(v1) << 16);
            }
            // ---- B: 32x128, 2048 pairs
            if (transB) {
                // pairs along K: contiguous float2, one ds_store_b32
#pragma unroll
                for (int it = 0; it < 8; ++it) {
                    int t  = tid + it * 256;
                    int nn = t >> 4;               // 0..127
                    int kk = (t & 15) << 1;
                    const float* p = B + (long)(bn + nn) * ldb + (k0 + kk);
                    float v0 = p[0], v1 = p[1];
                    int ct = nn >> 4, c = nn & 15;
                    int flane = ((kk >> 4) & 1) * 16 + c;
                    *(unsigned*)&Bsf[ct][flane][kk & 15] = bfbits(v0) | (bfbits(v1) << 16);
                }
            } else {
                // pairs along N: contiguous float2, two ds_store_b16 (adjacent lanes)
#pragma unroll
                for (int it = 0; it < 8; ++it) {
                    int t  = tid + it * 256;
                    int kk = t >> 6;               // 0..31
                    int nn = (t & 63) << 1;        // even column 0..126
                    const float* p = B + (long)(k0 + kk) * ldb + (bn + nn);
                    float v0 = p[0], v1 = p[1];
                    int ct = nn >> 4;              // nn even -> nn,nn+1 same 16-block
                    int c  = nn & 15;
                    int khi = (kk >> 4) & 1;
                    int e   = kk & 15;
                    Bsf[ct][khi * 16 + c][e]     = bfval(v0);
                    Bsf[ct][khi * 16 + c + 1][e] = bfval(v1);
                }
            }
        } else {
            // ---- guarded staging (M-edge blocks only in this workload)
            for (int t = tid; t < 1024; t += 256) {
                int rr = t >> 4;
                int cc = (t & 15) << 1;
                int gm = bm + rr, gk = k0 + cc;
                float v0 = 0.f, v1 = 0.f;
                if (gm < M) {
                    const float* p = A + (long)gm * lda + gk;
                    if (gk < K)     v0 = p[0];
                    if (gk + 1 < K) v1 = p[1];
                }
                int tile = rr >> 4, r = rr & 15;
                int flane = ((cc >> 4) & 1) * 16 + r;
                int e     = cc & 15;
                *(unsigned*)&Asf[tile][flane][e] = bfbits(v0) | (bfbits(v1) << 16);
            }
            for (int t = tid; t < 2048; t += 256) {
                int nn = t >> 4;
                int kk = (t & 15) << 1;
                int gk = k0 + kk, gn = bn + nn;
                float v0 = 0.f, v1 = 0.f;
                if (gn < N) {
                    if (transB) {
                        const float* p = B + (long)gn * ldb + gk;
                        if (gk < K)     v0 = p[0];
                        if (gk + 1 < K) v1 = p[1];
                    } else {
                        if (gk < K)     v0 = B[(long)gk * ldb + gn];
                        if (gk + 1 < K) v1 = B[(long)(gk + 1) * ldb + gn];
                    }
                }
                int ct = nn >> 4, c = nn & 15;
                int flane = ((kk >> 4) & 1) * 16 + c;
                *(unsigned*)&Bsf[ct][flane][kk & 15] = bfbits(v0) | (bfbits(v1) << 16);
            }
        }

        // prefetch next K-tile of A into cache (global_prefetch_b8)
        if (k0 + BK < K) {
            int rr = tid >> 2;                  // 64 rows, 4 threads/row
            int gm = bm + rr;
            if (gm < M)
                __builtin_prefetch(A + (long)gm * lda + (k0 + BK) + ((tid & 3) << 3), 0, 1);
        }

        __syncthreads();

        // fragment loads: contiguous 32B per lane -> ds_load_b128 x2 each
        v16bf a  = *(const v16bf*)(&Asf[art][lane][0]);
        v16bf b0 = *(const v16bf*)(&Bsf[bct0 + 0][lane][0]);
        v16bf b1 = *(const v16bf*)(&Bsf[bct0 + 1][lane][0]);
        v16bf b2 = *(const v16bf*)(&Bsf[bct0 + 2][lane][0]);
        v16bf b3 = *(const v16bf*)(&Bsf[bct0 + 3][lane][0]);

        acc[0] = __builtin_amdgcn_wmma_f32_16x16x32_bf16(false, a, false, b0,
                                                         (short)0, acc[0], false, false);
        acc[1] = __builtin_amdgcn_wmma_f32_16x16x32_bf16(false, a, false, b1,
                                                         (short)0, acc[1], false, false);
        acc[2] = __builtin_amdgcn_wmma_f32_16x16x32_bf16(false, a, false, b2,
                                                         (short)0, acc[2], false, false);
        acc[3] = __builtin_amdgcn_wmma_f32_16x16x32_bf16(false, a, false, b3,
                                                         (short)0, acc[3], false, false);
        __syncthreads();
    }

    // C/D layout: VGPR v -> row (lane>>4)*8+v, col = lane&15
    const int r     = lane & 15;
    const int crow0 = art * 16 + (lane >> 4) * 8;
#pragma unroll
    for (int ct = 0; ct < 4; ++ct) {
        int gn = bn + (bct0 + ct) * 16 + r;
        if (gn >= N) continue;
        float bi = bias ? bias[gn] : 0.f;
#pragma unroll
        for (int v = 0; v < 8; ++v) {
            int gm = bm + crow0 + v;
            if (gm < M) C[(long)gm * ldc + gn] = acc[ct][v] * scale + bi;
        }
    }
}

// ---------------- small kernels ----------------
__global__ void k_fill(float* p, float v, long n) {
    long i = (long)blockIdx.x * blockDim.x + threadIdx.x;
    if (i < n) p[i] = v;
}

__global__ void k_bcast_bias(float* p, const float* __restrict__ b, long M, int N) {
    long i = (long)blockIdx.x * blockDim.x + threadIdx.x;
    if (i < M * (long)N) p[i] = b[i % N];
}

// gate[i] = h_q[i,:] . gate_w + gate_b
__global__ void k_gate(const float* __restrict__ h_q, const float* __restrict__ gw,
                       const float* __restrict__ gb, float* __restrict__ gate, int nq) {
    int i = blockIdx.x * blockDim.x + threadIdx.x;
    if (i >= nq) return;
    float s = 0.f;
    for (int d = 0; d < HD_; ++d) s += h_q[i * HD_ + d] * gw[d];
    gate[i] = s + gb[0];
}

// attentional aggregation pooling: one block per graph b
__global__ __launch_bounds__(256)
void k_pool(const float* __restrict__ h_q, const float* __restrict__ gate,
            const int* __restrict__ qb, float* __restrict__ qpool, int nq) {
    int b = blockIdx.x;
    int tid = threadIdx.x;
    __shared__ float red[256];

    float m = -1e30f;
    for (int i = tid; i < nq; i += 256) if (qb[i] == b) m = fmaxf(m, gate[i]);
    red[tid] = m; __syncthreads();
    for (int s = 128; s > 0; s >>= 1) { if (tid < s) red[tid] = fmaxf(red[tid], red[tid + s]); __syncthreads(); }
    float gmax = red[0]; __syncthreads();

    float sum = 0.f;
    for (int i = tid; i < nq; i += 256) if (qb[i] == b) sum += __expf(gate[i] - gmax);
    red[tid] = sum; __syncthreads();
    for (int s = 128; s > 0; s >>= 1) { if (tid < s) red[tid] += red[tid + s]; __syncthreads(); }
    float gsum = red[0] + 1e-16f; __syncthreads();

    if (tid < HD_) {
        float acc = 0.f;
        for (int i = 0; i < nq; ++i)
            if (qb[i] == b) acc += (__expf(gate[i] - gmax) / gsum) * h_q[i * HD_ + tid];
        qpool[b * HD_ + tid] = acc;
    }
}

// column mean / biased var: one block per column
__global__ __launch_bounds__(256)
void k_colstats(const float* __restrict__ X, float* __restrict__ mean, float* __restrict__ var,
                int M, int N) {
    int c = blockIdx.x;
    __shared__ float s1[256], s2[256];
    float a = 0.f, b = 0.f;
    for (int r = threadIdx.x; r < M; r += 256) {
        float v = X[(long)r * N + c];
        a += v; b += v * v;
    }
    s1[threadIdx.x] = a; s2[threadIdx.x] = b; __syncthreads();
    for (int s = 128; s > 0; s >>= 1) {
        if (threadIdx.x < s) { s1[threadIdx.x] += s1[threadIdx.x + s]; s2[threadIdx.x] += s2[threadIdx.x + s]; }
        __syncthreads();
    }
    if (threadIdx.x == 0) {
        float m = s1[0] / (float)M;
        mean[c] = m;
        var[c]  = s2[0] / (float)M - m * m;
    }
}

// Y = elu(g*(X-mean)*rsqrt(var+1e-5)+be) (+res). In-place safe.
__global__ void k_bn_elu(const float* __restrict__ X, const float* __restrict__ mean,
                         const float* __restrict__ var, const float* __restrict__ g,
                         const float* __restrict__ be, const float* __restrict__ res,
                         float* __restrict__ Y, long M, int N) {
    long i = (long)blockIdx.x * blockDim.x + threadIdx.x;
    if (i >= M * (long)N) return;
    int c = (int)(i % N);
    float y = g[c] * (X[i] - mean[c]) * rsqrtf(var[c] + 1e-5f) + be[c];
    y = y > 0.f ? y : (__expf(y) - 1.f);
    if (res) y += res[i];
    Y[i] = y;
}

// alpha_src[i,h] = sum_c x1[i,h,c] * q2[batch[i], h*256 + c]
__global__ void k_alpha_src(const float* __restrict__ x1, const float* __restrict__ q2,
                            const int* __restrict__ batch, float* __restrict__ asrc, int n) {
    int idx = blockIdx.x * blockDim.x + threadIdx.x;
    if (idx >= n * H_) return;
    int i = idx >> 3, h = idx & 7;
    const float4* xv = (const float4*)(x1 + (long)i * (H_ * HD_) + h * HD_);
    const float4* av = (const float4*)(q2 + (long)batch[i] * (H_ * 2 * HD_) + h * (2 * HD_));
    float s = 0.f;
    for (int c = 0; c < HD_ / 4; ++c) {
        float4 x = xv[c], a = av[c];
        s += x.x * a.x + x.y * a.y + x.z * a.z + x.w * a.w;
    }
    asrc[idx] = s;
}

__device__ __forceinline__ float lrelu02(float x) { return x > 0.f ? x : 0.2f * x; }

__global__ void k_edge_max(const int* __restrict__ ei, const float* __restrict__ asrc,
                           float* __restrict__ dmax, int E) {
    int idx = blockIdx.x * blockDim.x + threadIdx.x;
    if (idx >= E * H_) return;
    int e = idx >> 3, h = idx & 7;
    int s = ei[e], d = ei[E + e];
    atomicMaxFloat(&dmax[d * H_ + h], lrelu02(asrc[s * H_ + h]));
}

__global__ void k_edge_exp(const int* __restrict__ ei, const float* __restrict__ asrc,
                           const float* __restrict__ dmax, float* __restrict__ dsum,
                           float* __restrict__ eexp, int E) {
    int idx = blockIdx.x * blockDim.x + threadIdx.x;
    if (idx >= E * H_) return;
    int e = idx >> 3, h = idx & 7;
    int s = ei[e], d = ei[E + e];
    float ex = __expf(lrelu02(asrc[s * H_ + h]) - dmax[d * H_ + h]);
    eexp[idx] = ex;
    atomicAdd(&dsum[d * H_ + h], ex);
}

__global__ void k_edge_norm(const int* __restrict__ ei, const float* __restrict__ dsum,
                            float* __restrict__ alpha, int E) {
    int idx = blockIdx.x * blockDim.x + threadIdx.x;
    if (idx >= E * H_) return;
    int e = idx >> 3, h = idx & 7;
    int d = ei[E + e];
    alpha[idx] = alpha[idx] / (dsum[d * H_ + h] + 1e-16f);
}

// out[dst, h*128+c] += alpha[e,h] * x1[src, h*128+c]; one block per edge
__global__ __launch_bounds__(256)
void k_edge_agg(const int* __restrict__ ei, const float* __restrict__ x1,
                const float* __restrict__ alpha, float* __restrict__ out, int E) {
    int e = blockIdx.x;
    if (e >= E) return;
    int s = ei[e], d = ei[E + e];
    for (int t = threadIdx.x; t < H_ * HD_; t += 256) {
        int h = t >> 7;
        atomicAdd(&out[(long)d * (H_ * HD_) + t], alpha[e * H_ + h] * x1[(long)s * (H_ * HD_) + t]);
    }
}

// masked row softmax over logits derived from sim (in place)
__global__ __launch_bounds__(256)
void k_row_softmax(float* __restrict__ sim, const unsigned char* __restrict__ mask,
                   const float* __restrict__ tau, int rows, int cols) {
    int r = blockIdx.x;
    if (r >= rows) return;
    __shared__ float red[256];
    const float invsig = 1.f + __expf(-tau[0]);   // 1/sigmoid(tau)
    const long base = (long)r * cols;

    float lmax = -1e30f;
    for (int c = threadIdx.x; c < cols; c += 256) {
        float l = mask[base + c] ? sim[base + c] * invsig : -1e9f;
        lmax = fmaxf(lmax, l);
    }
    red[threadIdx.x] = lmax; __syncthreads();
    for (int s = 128; s > 0; s >>= 1) { if (threadIdx.x < s) red[threadIdx.x] = fmaxf(red[threadIdx.x], red[threadIdx.x + s]); __syncthreads(); }
    float rmax = red[0]; __syncthreads();

    float lsum = 0.f;
    for (int c = threadIdx.x; c < cols; c += 256) {
        float l = mask[base + c] ? sim[base + c] * invsig : -1e9f;
        lsum += __expf(l - rmax);
    }
    red[threadIdx.x] = lsum; __syncthreads();
    for (int s = 128; s > 0; s >>= 1) { if (threadIdx.x < s) red[threadIdx.x] += red[threadIdx.x + s]; __syncthreads(); }
    float rsum = red[0]; __syncthreads();

    for (int c = threadIdx.x; c < cols; c += 256) {
        float l = mask[base + c] ? sim[base + c] * invsig : -1e9f;
        sim[base + c] = __expf(l - rmax) / rsum;
    }
}

// ---------------- host launcher ----------------
static inline void launch_gemm(const float* A, const float* B, const float* bias, float* C,
                               int M, int N, int K, int lda, int ldb, int ldc,
                               int transB, float scale, hipStream_t stream) {
    dim3 g((N + BN - 1) / BN, (M + BM - 1) / BM);
    gemm_bf16_wmma<<<g, 256, 0, stream>>>(A, B, bias, C, M, N, K, lda, ldb, ldc, transB, scale);
}

extern "C" void kernel_launch(void* const* d_in, const int* in_sizes, int n_in,
                              void* d_out, int out_size, void* d_ws, size_t ws_size,
                              hipStream_t stream) {
    const float* h_t   = (const float*)d_in[0];
    const float* h_q   = (const float*)d_in[1];
    const float* mm    = (const float*)d_in[2];
    const unsigned char* mask = (const unsigned char*)d_in[5];
    const int* tei     = (const int*)d_in[6];
    const int* qei     = (const int*)d_in[7];
    const int* tb      = (const int*)d_in[8];
    const int* qb      = (const int*)d_in[9];
    const float* tau   = (const float*)d_in[10];
    const float* gate_w = (const float*)d_in[11];
    const float* gate_b = (const float*)d_in[12];
    const float* m0w1 = (const float*)d_in[13]; const float* m0b1 = (const float*)d_in[14];
    const float* m0g1 = (const float*)d_in[15]; const float* m0e1 = (const float*)d_in[16];
    const float* m0w2 = (const float*)d_in[17]; const float* m0b2 = (const float*)d_in[18];
    const float* m0g2 = (const float*)d_in[19]; const float* m0e2 = (const float*)d_in[20];
    const float* gatw = (const float*)d_in[21]; const float* gatb = (const float*)d_in[22];
    const float* m1w1 = (const float*)d_in[23]; const float* m1b1 = (const float*)d_in[24];
    const float* m1g1 = (const float*)d_in[25]; const float* m1e1 = (const float*)d_in[26];
    const float* m1w2 = (const float*)d_in[27]; const float* m1b2 = (const float*)d_in[28];
    const float* m1g2 = (const float*)d_in[29]; const float* m1e2 = (const float*)d_in[30];

    // output layout: h_t2 | h_q2 | a_t | a_q | mm_out
    float* out   = (float*)d_out;
    float* ht2   = out;
    float* hq2   = out + (long)NT_ * HD_;
    float* a_t   = hq2 + (long)NQ_ * HD_;
    float* a_q   = a_t + (long)ET_ * H_;
    float* mmout = a_q + (long)EQ_ * H_;

    // workspace carve-up
    float* w = (float*)d_ws;
    size_t o = 0;
    auto alloc = [&](size_t n) { float* p = w + o; o += n; return p; };
    float* buf_n    = alloc((size_t)NQ_ * HD_);
    float* buf_gate = alloc(NQ_);
    float* buf_qp   = alloc((size_t)B_ * HD_);
    float* buf_pre1 = alloc((size_t)B_ * 256);
    float* buf_act1 = alloc((size_t)B_ * 256);
    float* buf_q2   = alloc((size_t)B_ * 2048);
    float* buf_x1t  = alloc((size_t)NT_ * H_ * HD_);
    float* buf_x1q  = alloc((size_t)NQ_ * H_ * HD_);
    float* asrct    = alloc((size_t)NT_ * H_);
    float* asrcq    = alloc((size_t)NQ_ * H_);
    float* dmax_t   = alloc((size_t)NT_ * H_);
    float* dsum_t   = alloc((size_t)NT_ * H_);
    float* dmax_q   = alloc((size_t)NQ_ * H_);
    float* dsum_q   = alloc((size_t)NQ_ * H_);
    float* buf_gt   = alloc((size_t)NT_ * H_ * HD_);
    float* buf_gq   = alloc((size_t)NQ_ * H_ * HD_);
    float* buf_tmid = alloc((size_t)NT_ * 256);
    float* buf_tout = alloc((size_t)NT_ * 128);
    float* buf_qmid = alloc((size_t)NQ_ * 256);
    float* buf_qout = alloc((size_t)NQ_ * 128);
    float* stat_m   = alloc(2048);
    float* stat_v   = alloc(2048);

    auto blks = [](long n) { return (unsigned)((n + 255) / 256); };

    // 1) n = mm @ h_t                        [NQ,HD] (HBM-bound: mm is 256MB)
    launch_gemm(mm, h_t, nullptr, buf_n, NQ_, HD_, NT_, NT_, HD_, HD_, 0, 1.f, stream);

    // 2) gate + attentional pooling -> q_pool [B,HD]
    k_gate<<<blks(NQ_), 256, 0, stream>>>(h_q, gate_w, gate_b, buf_gate, NQ_);
    k_pool<<<B_, 256, 0, stream>>>(h_q, buf_gate, qb, buf_qp, NQ_);

    // 3) MLP0: [B,128] -> [B,256] -> [B,2048]
    launch_gemm(buf_qp, m0w1, m0b1, buf_pre1, B_, 256, 128, 128, 256, 256, 0, 1.f, stream);
    k_colstats<<<256, 256, 0, stream>>>(buf_pre1, stat_m, stat_v, B_, 256);
    k_bn_elu<<<blks((long)B_ * 256), 256, 0, stream>>>(buf_pre1, stat_m, stat_v, m0g1, m0e1, nullptr, buf_act1, B_, 256);
    launch_gemm(buf_act1, m0w2, m0b2, buf_q2, B_, 2048, 256, 256, 2048, 2048, 0, 1.f, stream);
    k_colstats<<<2048, 256, 0, stream>>>(buf_q2, stat_m, stat_v, B_, 2048);
    k_bn_elu<<<blks((long)B_ * 2048), 256, 0, stream>>>(buf_q2, stat_m, stat_v, m0g2, m0e2, nullptr, buf_q2, B_, 2048);

    // 4) x1 = x @ gat_w (no bias here; bias folded into aggregation init)
    launch_gemm(h_t,   gatw, nullptr, buf_x1t, NT_, H_ * HD_, HD_, HD_, H_ * HD_, H_ * HD_, 0, 1.f, stream);
    launch_gemm(buf_n, gatw, nullptr, buf_x1q, NQ_, H_ * HD_, HD_, HD_, H_ * HD_, H_ * HD_, 0, 1.f, stream);

    // 5) per-node attention logits
    k_alpha_src<<<blks((long)NT_ * H_), 256, 0, stream>>>(buf_x1t, buf_q2, tb, asrct, NT_);
    k_alpha_src<<<blks((long)NQ_ * H_), 256, 0, stream>>>(buf_x1q, buf_q2, qb, asrcq, NQ_);

    // 6) target-edge softmax + aggregation
    k_fill<<<blks((long)NT_ * H_), 256, 0, stream>>>(dmax_t, -1e30f, (long)NT_ * H_);
    k_fill<<<blks((long)NT_ * H_), 256, 0, stream>>>(dsum_t, 0.f, (long)NT_ * H_);
    k_bcast_bias<<<blks((long)NT_ * H_ * HD_), 256, 0, stream>>>(buf_gt, gatb, NT_, H_ * HD_);
    k_edge_max<<<blks((long)ET_ * H_), 256, 0, stream>>>(tei, asrct, dmax_t, ET_);
    k_edge_exp<<<blks((long)ET_ * H_), 256, 0, stream>>>(tei, asrct, dmax_t, dsum_t, a_t, ET_);
    k_edge_norm<<<blks((long)ET_ * H_), 256, 0, stream>>>(tei, dsum_t, a_t, ET_);
    k_edge_agg<<<ET_, 256, 0, stream>>>(tei, buf_x1t, a_t, buf_gt, ET_);

    // 7) query-edge softmax + aggregation
    k_fill<<<blks((long)NQ_ * H_), 256, 0, stream>>>(dmax_q, -1e30f, (long)NQ_ * H_);
    k_fill<<<blks((long)NQ_ * H_), 256, 0, stream>>>(dsum_q, 0.f, (long)NQ_ * H_);
    k_bcast_bias<<<blks((long)NQ_ * H_ * HD_), 256, 0, stream>>>(buf_gq, gatb, NQ_, H_ * HD_);
    k_edge_max<<<blks((long)EQ_ * H_), 256, 0, stream>>>(qei, asrcq, dmax_q, EQ_);
    k_edge_exp<<<blks((long)EQ_ * H_), 256, 0, stream>>>(qei, asrcq, dmax_q, dsum_q, a_q, EQ_);
    k_edge_norm<<<blks((long)EQ_ * H_), 256, 0, stream>>>(qei, dsum_q, a_q, EQ_);
    k_edge_agg<<<EQ_, 256, 0, stream>>>(qei, buf_x1q, a_q, buf_gq, EQ_);

    // 8) MLP1 target branch (+ residual h_t) -> ht2
    launch_gemm(buf_gt, m1w1, m1b1, buf_tmid, NT_, 256, 1024, 1024, 256, 256, 0, 1.f, stream);
    k_colstats<<<256, 256, 0, stream>>>(buf_tmid, stat_m, stat_v, NT_, 256);
    k_bn_elu<<<blks((long)NT_ * 256), 256, 0, stream>>>(buf_tmid, stat_m, stat_v, m1g1, m1e1, nullptr, buf_tmid, NT_, 256);
    launch_gemm(buf_tmid, m1w2, m1b2, buf_tout, NT_, 128, 256, 256, 128, 128, 0, 1.f, stream);
    k_colstats<<<128, 256, 0, stream>>>(buf_tout, stat_m, stat_v, NT_, 128);
    k_bn_elu<<<blks((long)NT_ * 128), 256, 0, stream>>>(buf_tout, stat_m, stat_v, m1g2, m1e2, h_t, ht2, NT_, 128);

    // 9) MLP1 query branch (+ residual h_q) -> hq2
    launch_gemm(buf_gq, m1w1, m1b1, buf_qmid, NQ_, 256, 1024, 1024, 256, 256, 0, 1.f, stream);
    k_colstats<<<256, 256, 0, stream>>>(buf_qmid, stat_m, stat_v, NQ_, 256);
    k_bn_elu<<<blks((long)NQ_ * 256), 256, 0, stream>>>(buf_qmid, stat_m, stat_v, m1g1, m1e1, nullptr, buf_qmid, NQ_, 256);
    launch_gemm(buf_qmid, m1w2, m1b2, buf_qout, NQ_, 128, 256, 256, 128, 128, 0, 1.f, stream);
    k_colstats<<<128, 256, 0, stream>>>(buf_qout, stat_m, stat_v, NQ_, 128);
    k_bn_elu<<<blks((long)NQ_ * 128), 256, 0, stream>>>(buf_qout, stat_m, stat_v, m1g2, m1e2, h_q, hq2, NQ_, 128);

    // 10) sim = (h_q2 @ h_t2^T) / sqrt(128); masked softmax -> mm_out
    launch_gemm(hq2, ht2, nullptr, mmout, NQ_, NT_, HD_, HD_, HD_, NT_, 1, 0.08838834764831845f, stream);
    k_row_softmax<<<NQ_, 256, 0, stream>>>(mmout, mask, tau, NQ_, NT_);
}
